// V2VNet_90323162235712
// MI455X (gfx1250) — matrix operand order repaired and checked by reading, not compiled, not run on hardware
//
#include <hip/hip_runtime.h>
#include <hip/hip_bf16.h>

typedef __attribute__((ext_vector_type(16))) _Float16 v16h;
typedef __attribute__((ext_vector_type(8)))  _Float16 v8h;
typedef __attribute__((ext_vector_type(8)))  float    v8f;

#define WMMA_F32_F16(a, b, c) \
  __builtin_amdgcn_wmma_f32_16x16x32_f16(false, (a), false, (b), (short)0, (c), false, false)

#define SHUF16(lo, hh) __builtin_shufflevector((lo), (hh), 0, 1, 2, 3, 4, 5, 6, 7, \
                                               8, 9, 10, 11, 12, 13, 14, 15)

// ---------------------------------------------------------------------------
// NCHW f32 -> NHWC f16 with channel padding (zeros for c >= C).
// grid = (cdiv(HW*CP,256), 1, nImg)
// ---------------------------------------------------------------------------
__global__ void nchw_to_nhwc16(const float* __restrict__ src, _Float16* __restrict__ dst,
                               int C, int CP, int HW, int imgOutBase) {
  int t = blockIdx.x * blockDim.x + threadIdx.x;
  if (t >= HW * CP) return;
  int z = blockIdx.z;
  int p = t / CP, c = t - (t / CP) * CP;
  _Float16 v = (_Float16)0.f;
  if (c < C) v = (_Float16)src[((long)z * C + c) * HW + p];
  dst[((long)(imgOutBase + z) * HW + p) * CP + c] = v;
}

// ---------------------------------------------------------------------------
// Pack GEMM B [Kreal][Nreal] f32 into WMMA fragment order (f16):
//   dst[(((nt*KT + kt)*32 + lane)*16 + e], lane = khalf*16 + n,
//   k = kt*32 + khalf*8 + (e&7) + (e>>3)*16, ncol = nt*16 + n.   (K%32==0, N%16==0 here)
// ---------------------------------------------------------------------------
__global__ void pack_b_frag(const float* __restrict__ src, _Float16* __restrict__ dst,
                            int Kreal, int Nreal) {
  int t = blockIdx.x * blockDim.x + threadIdx.x;
  if (t >= Kreal * Nreal) return;
  int KT   = Kreal >> 5;
  int e    = t & 15;
  int lane = (t >> 4) & 31;
  int kt   = (t >> 9) % KT;
  int nt   = t / (KT << 9);
  int k    = kt * 32 + (lane >> 4) * 8 + (e & 7) + (e >> 3) * 16;
  int ncol = nt * 16 + (lane & 15);
  dst[t] = (_Float16)src[(long)k * Nreal + ncol];
}

// ---------------------------------------------------------------------------
// Pack conv weight [O][Cin][3][3] f32 into fragment order with K ordered
// (r,s)-major / channel-minor over padded channels:  k = rs*CPAD + c.
// Zero for rs>=9, c>=Cin, ncol>=O.
// ---------------------------------------------------------------------------
__global__ void pack_conv_frag(const float* __restrict__ w, _Float16* __restrict__ dst,
                               int Cin, int CPAD, int O, int Kpad, int Npad) {
  int t = blockIdx.x * blockDim.x + threadIdx.x;
  if (t >= Kpad * Npad) return;
  int KT   = Kpad >> 5;
  int e    = t & 15;
  int lane = (t >> 4) & 31;
  int kt   = (t >> 9) % KT;
  int nt   = t / (KT << 9);
  int k    = kt * 32 + (lane >> 4) * 8 + (e & 7) + (e >> 3) * 16;
  int ncol = nt * 16 + (lane & 15);
  int rs = k / CPAD, c = k - (k / CPAD) * CPAD;
  float v = 0.f;
  if (rs < 9 && c < Cin && ncol < O) v = w[((long)ncol * Cin + c) * 9 + rs];
  dst[t] = (_Float16)v;
}

__global__ void cvt_f32_f16(const float* __restrict__ s, _Float16* __restrict__ d, long n) {
  long t = (long)blockIdx.x * blockDim.x + threadIdx.x;
  if (t < n) d[t] = (_Float16)s[t];
}

// ---------------------------------------------------------------------------
// Implicit-GEMM 3x3 VALID conv + bias + ReLU on NHWC f16 activations.
// One wave computes 16 pixels x (16*NT) channels.
// K order = (r,s)-major, channel-minor:  a 32-k tile is a contiguous channel
// run at one spatial tap (or two taps for CPAD==16), so the A fragment is two
// 16-byte vector loads per lane per k-step. No predication anywhere: addresses
// are clamped (garbage rows only affect unstored output rows); padded
// channels/taps are zeros in buffers/weights.
// grid = (ceil(HWout/16), Npad/(16*NT), img)
// ---------------------------------------------------------------------------
template <int CPAD, int NT>
__global__ void conv3x3_relu_wmma(const _Float16* __restrict__ in,  // NHWC [nImg][Hin][Win][CPAD]
                                  const _Float16* __restrict__ Bp,  // frag-packed weights
                                  const float* __restrict__ bias,   // [Cout]
                                  _Float16* __restrict__ out16,     // NHWC [64][HWout][Cout] or null
                                  float* __restrict__ out32,        // NCHW [64][Cout][HWout] or null
                                  int Hin, int Win, int Hout, int Wout,
                                  int Cout, int stride, int KT, int imgOutBase) {
  const int  lane = threadIdx.x;
  const bool hi   = lane >= 16;
  const int  mn   = lane & 15;
  const int  HWout = Hout * Wout;
  const int  imgIn = blockIdx.z;
  const int  img   = imgOutBase + imgIn;

  const int p   = blockIdx.x * 16 + mn;
  const int pcl = (p < HWout) ? p : (HWout - 1);      // clamp -> always-safe address
  const int oy  = pcl / Wout, ox = pcl - (pcl / Wout) * Wout;
  const _Float16* pix = in + ((long)imgIn * Hin * Win + (long)(oy * stride) * Win + ox * stride)
                            * CPAD + (hi ? 8 : 0);
  const _Float16* BpL = Bp + ((long)blockIdx.y * NT * KT * 32 + lane) * 16;

  constexpr int G = (CPAD >= 32) ? (CPAD / 32) : 1;   // 32-k tiles per spatial tap

  v8f acc[NT] = {};
  for (int kt = 0; kt < KT; ++kt) {
    v16h a;
    if (CPAD >= 32) {
      int rs = kt / G;
      int c0 = (kt - rs * G) * 32;
      int r = rs / 3, s = rs - (rs / 3) * 3;
      const _Float16* q = pix + (r * Win + s) * CPAD + c0;
      v8h lo = *(const v8h*)(q);
      v8h hh = *(const v8h*)(q + 16);
      a = SHUF16(lo, hh);
    } else {                                          // CPAD == 16: two taps per tile
      int rs0 = kt * 2, rs1 = rs0 + 1;
      int r0 = rs0 / 3, s0 = rs0 - r0 * 3;
      int r1 = rs1 / 3, s1 = rs1 - r1 * 3;
      v8h lo = *(const v8h*)(pix + (r0 * Win + s0) * CPAD);
      v8h hh = *(const v8h*)(pix + (r1 * Win + s1) * CPAD);
      a = SHUF16(lo, hh);
    }
#pragma unroll
    for (int nt = 0; nt < NT; ++nt) {
      v16h b = *(const v16h*)(BpL + ((long)(nt * KT + kt)) * 512);
      acc[nt] = WMMA_F32_F16(a, b, acc[nt]);
    }
  }

#pragma unroll
  for (int nt = 0; nt < NT; ++nt) {
    int ocol = (blockIdx.y * NT + nt) * 16 + mn;
    if (ocol < Cout) {
      float bb = bias[ocol];
#pragma unroll
      for (int v = 0; v < 8; ++v) {
        int pp = blockIdx.x * 16 + v + (hi ? 8 : 0);
        if (pp < HWout) {
          float x = fmaxf(acc[nt][v] + bb, 0.f);
          if (out16) out16[((long)img * HWout + pp) * Cout + ocol] = (_Float16)x;  // NHWC
          if (out32) out32[((long)img * Cout + ocol) * HWout + pp] = x;            // NCHW
        }
      }
    }
  }
}

// ---------------------------------------------------------------------------
// f16 GEMM, A row-major [M,K] (M,N multiples of 16, K%32==0), B frag-packed.
// ---------------------------------------------------------------------------
template <int NT>
__global__ void gemm_wmma(const _Float16* __restrict__ A, const _Float16* __restrict__ Bp,
                          float* __restrict__ C, int M, int K, int N) {
  const int  lane = threadIdx.x;
  const bool hi   = lane >= 16;
  const int  mn   = lane & 15;
  const int  mrow = blockIdx.x * 16 + mn;
  const int  KT   = K >> 5;
  const _Float16* Arow = A + (long)mrow * K + (hi ? 8 : 0);
  const _Float16* BpL  = Bp + ((long)blockIdx.y * NT * KT * 32 + lane) * 16;

  v8f acc[NT] = {};
  for (int kt = 0; kt < KT; ++kt) {
    v8h lo = *(const v8h*)(Arow + kt * 32);
    v8h hh = *(const v8h*)(Arow + kt * 32 + 16);
    v16h a = SHUF16(lo, hh);
#pragma unroll
    for (int nt = 0; nt < NT; ++nt) {
      v16h b = *(const v16h*)(BpL + ((long)(nt * KT + kt)) * 512);
      acc[nt] = WMMA_F32_F16(a, b, acc[nt]);
    }
  }
#pragma unroll
  for (int nt = 0; nt < NT; ++nt) {
    int ncol = (blockIdx.y * NT + nt) * 16 + mn;
#pragma unroll
    for (int v = 0; v < 8; ++v) {
      int m = blockIdx.x * 16 + v + (hi ? 8 : 0);
      C[(long)m * N + ncol] = acc[nt][v];
    }
  }
}

// ---------------------------------------------------------------------------
// Bilinear affine warp (grid_sample, zeros padding, align_corners=False).
// act3 is NCHW f32.
// ---------------------------------------------------------------------------
__global__ void warp_sample(const float* __restrict__ act3,  // [64][4][68*68]
                            const float* __restrict__ tf,    // [16][3][4][4]
                            _Float16* __restrict__ xn) {     // [64][18496]
  const int HW = 68;
  int t = blockIdx.x * blockDim.x + threadIdx.x;
  if (t >= 16 * 3 * HW * HW) return;
  int w = t % HW;
  int h = (t / HW) % HW;
  int i = (t / (HW * HW)) % 3;
  int b = t / (3 * HW * HW);

  const float* T = tf + ((b * 3 + i) << 4);
  float gx = (2.f * w + 1.f) / 68.f - 1.f;
  float gy = (2.f * h + 1.f) / 68.f - 1.f;
  float xg = T[0] * gx + T[1] * gy + T[3];
  float yg = T[4] * gx + T[5] * gy + T[7];
  float ixf = ((xg + 1.f) * 68.f - 1.f) * 0.5f;
  float iyf = ((yg + 1.f) * 68.f - 1.f) * 0.5f;
  float x0f = floorf(ixf), y0f = floorf(iyf);
  int   x0 = (int)x0f, y0 = (int)y0f;
  float wx1 = ixf - x0f, wy1 = iyf - y0f;
  float w00 = (1.f - wx1) * (1.f - wy1), w10 = wx1 * (1.f - wy1);
  float w01 = (1.f - wx1) * wy1,         w11 = wx1 * wy1;

  int img = 16 + b * 3 + i;
  const float* base = act3 + (long)img * 4 * HW * HW;
  long row = (long)((i + 1) * 16 + b) * 18496;
#pragma unroll
  for (int c = 0; c < 4; ++c) {
    const float* ch = base + c * HW * HW;
    auto tap = [&](int xi, int yi) -> float {
      return (xi >= 0 && xi < HW && yi >= 0 && yi < HW) ? ch[yi * HW + xi] : 0.f;
    };
    float val = tap(x0, y0) * w00 + tap(x0 + 1, y0) * w10 +
                tap(x0, y0 + 1) * w01 + tap(x0 + 1, y0 + 1) * w11;
    xn[row + c * HW * HW + h * HW + w] = (_Float16)val;
  }
}

// ---------------------------------------------------------------------------
// GCN max-aggregation: deg==4 everywhere => every edge norm == 0.25, so
//   agg[any node of sample b] = max_j 0.25 * xw[j*16+b]  (+bias, opt ReLU).
// ---------------------------------------------------------------------------
__global__ void gcn_combine(const float* __restrict__ xw, const float* __restrict__ bias,
                            _Float16* __restrict__ out16, float* __restrict__ pooled,
                            int doRelu) {
  int t = blockIdx.x * blockDim.x + threadIdx.x;
  if (t >= 16 * 256) return;
  int b = t >> 8, f = t & 255;
  float v = -__builtin_inff();
#pragma unroll
  for (int j = 0; j < 4; ++j) v = fmaxf(v, 0.25f * xw[((j * 16 + b) << 8) + f]);
  v += bias[f];
  if (doRelu) v = fmaxf(v, 0.f);
  if (out16) {
#pragma unroll
    for (int j = 0; j < 4; ++j) out16[((j * 16 + b) << 8) + f] = (_Float16)v;
  }
  if (pooled) pooled[t] = v;
}

// ---------------------------------------------------------------------------
// Head: speed MLP (1->16->8) ++ pooled -> FC 264->256->64->3, single block.
// Output = (pc[:,0] | pc[:,1] | pc[:,2]) = out[col*16 + b].
// ---------------------------------------------------------------------------
__global__ void head_mlp(const float* __restrict__ pooled, const float* __restrict__ speed,
                         const float* __restrict__ mw1, const float* __restrict__ mb1,
                         const float* __restrict__ mw2, const float* __restrict__ mb2,
                         const float* __restrict__ fw1, const float* __restrict__ fb1,
                         const float* __restrict__ fw2, const float* __restrict__ fb2,
                         const float* __restrict__ fw3, const float* __restrict__ fb3,
                         float* __restrict__ out) {
  __shared__ float G[16][264];
  __shared__ float H1[16][256];
  __shared__ float H2[16][64];
  int tid = threadIdx.x;
  for (int t = tid; t < 16 * 256; t += blockDim.x) G[t >> 8][t & 255] = pooled[t];
  if (tid < 16) {
    int b = tid;
    float m1[16];
#pragma unroll
    for (int j = 0; j < 16; ++j) m1[j] = fmaxf(speed[b] * mw1[j] + mb1[j], 0.f);
#pragma unroll
    for (int j = 0; j < 8; ++j) {
      float s = mb2[j];
      for (int q = 0; q < 16; ++q) s += m1[q] * mw2[q * 8 + j];
      G[b][256 + j] = fmaxf(s, 0.f);
    }
  }
  __syncthreads();
  for (int t = tid; t < 16 * 256; t += blockDim.x) {
    int b = t >> 8, o = t & 255;
    float s = fb1[o];
    for (int q = 0; q < 264; ++q) s += G[b][q] * fw1[q * 256 + o];
    H1[b][o] = fmaxf(s, 0.f);
  }
  __syncthreads();
  for (int t = tid; t < 16 * 64; t += blockDim.x) {
    int b = t >> 6, o = t & 63;
    float s = fb2[o];
    for (int q = 0; q < 256; ++q) s += H1[b][q] * fw2[q * 64 + o];
    H2[b][o] = fmaxf(s, 0.f);
  }
  __syncthreads();
  if (tid < 48) {
    int col = tid >> 4, b = tid & 15;
    float s = fb3[col];
    for (int q = 0; q < 64; ++q) s += H2[b][q] * fw3[q * 3 + col];
    out[tid] = s;
  }
}

// ---------------------------------------------------------------------------
static inline unsigned cdiv(unsigned a, unsigned b) { return (a + b - 1) / b; }

extern "C" void kernel_launch(void* const* d_in, const int* in_sizes, int n_in,
                              void* d_out, int out_size, void* d_ws, size_t ws_size,
                              hipStream_t stream) {
  (void)in_sizes; (void)n_in; (void)out_size; (void)ws_size;

  const float* ego_lidar   = (const float*)d_in[0];
  const float* ego_speed   = (const float*)d_in[1];
  const float* other_lidar = (const float*)d_in[2];
  const float* other_tf    = (const float*)d_in[3];
  const float* ew1 = (const float*)d_in[4];   const float* eb1 = (const float*)d_in[5];
  const float* ew2 = (const float*)d_in[6];   const float* eb2 = (const float*)d_in[7];
  const float* ew3 = (const float*)d_in[8];   const float* eb3 = (const float*)d_in[9];
  const float* ow1 = (const float*)d_in[10];  const float* ob1 = (const float*)d_in[11];
  const float* ow2 = (const float*)d_in[12];  const float* ob2 = (const float*)d_in[13];
  const float* ow3 = (const float*)d_in[14];  const float* ob3 = (const float*)d_in[15];
  const float* gw1 = (const float*)d_in[16];  const float* gb1 = (const float*)d_in[17];
  const float* gw2 = (const float*)d_in[18];  const float* gb2 = (const float*)d_in[19];
  const float* gw3 = (const float*)d_in[20];  const float* gb3 = (const float*)d_in[21];
  const float* fw1 = (const float*)d_in[22];  const float* fb1 = (const float*)d_in[23];
  const float* fw2 = (const float*)d_in[24];  const float* fb2 = (const float*)d_in[25];
  const float* fw3 = (const float*)d_in[26];  const float* fb3 = (const float*)d_in[27];
  const float* mw1 = (const float*)d_in[28];  const float* mb1 = (const float*)d_in[29];
  const float* mw2 = (const float*)d_in[30];  const float* mb2 = (const float*)d_in[31];

  // Geometry
  const int C0 = 10, C0p = 16, C1 = 128, C2 = 64, C3 = 4;
  const int H0 = 280, W0 = 280;  const int HW0 = H0 * W0;   // 78400
  const int H1 = 139, W1 = 139;  const int HW1 = H1 * W1;   // 19321
  const int H2 = 137, W2 = 137;  const int HW2 = H2 * W2;   // 18769
  const int H3 = 68,  W3 = 68;   const int HW3 = H3 * W3;   // 4624
  const int K1p = 160;                  // 10 rs-taps x 16 padded channels (rs=9 zero)
  const int K2  = 9 * C1;               // 1152
  const int K3  = 9 * C2;               // 576
  const int O1p = 128, O2p = 64, O3p = 16;
  const int FEAT = C3 * HW3;            // 18496 = 578*32
  const int NH = 256;

  // Workspace carve-out
  char* ws = (char*)d_ws;
  size_t off = 0;
  auto carve = [&](size_t bytes) -> void* {
    void* p = ws + off;
    off += (bytes + 255) & ~(size_t)255;
    return p;
  };
  _Float16* lid  = (_Float16*)carve((size_t)64 * HW0 * C0p * 2);  // NHWC lidar ~160 MB
  _Float16* act1 = (_Float16*)carve((size_t)64 * HW1 * C1 * 2);   // NHWC ~316 MB
  _Float16* act2 = (_Float16*)carve((size_t)64 * HW2 * C2 * 2);   // NHWC ~154 MB
  float*    act3 = (float*)   carve((size_t)64 * C3 * HW3 * 4);   // NCHW ~4.7 MB
  _Float16* wp1e = (_Float16*)carve((size_t)K1p * O1p * 2);
  _Float16* wp1o = (_Float16*)carve((size_t)K1p * O1p * 2);
  _Float16* wp2e = (_Float16*)carve((size_t)K2 * O2p * 2);
  _Float16* wp2o = (_Float16*)carve((size_t)K2 * O2p * 2);
  _Float16* wp3e = (_Float16*)carve((size_t)K3 * O3p * 2);
  _Float16* wp3o = (_Float16*)carve((size_t)K3 * O3p * 2);
  _Float16* gw1h = (_Float16*)carve((size_t)FEAT * NH * 2);
  _Float16* gw2h = (_Float16*)carve((size_t)NH * NH * 2);
  _Float16* gw3h = (_Float16*)carve((size_t)NH * NH * 2);
  _Float16* xn   = (_Float16*)carve((size_t)64 * FEAT * 2);
  float*    xw   = (float*)   carve((size_t)64 * NH * 4);
  _Float16* a16  = (_Float16*)carve((size_t)64 * NH * 2);
  float*    pooled = (float*) carve((size_t)16 * NH * 4);

  // --- Input relayout to NHWC f16 (channels padded 10->16) ---
  nchw_to_nhwc16<<<dim3(cdiv(HW0 * C0p, 256), 1, 16), 256, 0, stream>>>(
      ego_lidar, lid, C0, C0p, HW0, 0);
  nchw_to_nhwc16<<<dim3(cdiv(HW0 * C0p, 256), 1, 48), 256, 0, stream>>>(
      other_lidar, lid, C0, C0p, HW0, 16);

  // --- Weight packing into fragment order ---
  pack_conv_frag<<<cdiv(K1p * O1p, 256), 256, 0, stream>>>(ew1, wp1e, C0, C0p, C1, K1p, O1p);
  pack_conv_frag<<<cdiv(K1p * O1p, 256), 256, 0, stream>>>(ow1, wp1o, C0, C0p, C1, K1p, O1p);
  pack_conv_frag<<<cdiv(K2 * O2p, 256), 256, 0, stream>>>(ew2, wp2e, C1, C1, C2, K2, O2p);
  pack_conv_frag<<<cdiv(K2 * O2p, 256), 256, 0, stream>>>(ow2, wp2o, C1, C1, C2, K2, O2p);
  pack_conv_frag<<<cdiv(K3 * O3p, 256), 256, 0, stream>>>(ew3, wp3e, C2, C2, C3, K3, O3p);
  pack_conv_frag<<<cdiv(K3 * O3p, 256), 256, 0, stream>>>(ow3, wp3o, C2, C2, C3, K3, O3p);
  pack_b_frag<<<cdiv(FEAT * NH, 256), 256, 0, stream>>>(gw1, gw1h, FEAT, NH);
  pack_b_frag<<<cdiv(NH * NH, 256), 256, 0, stream>>>(gw2, gw2h, NH, NH);
  pack_b_frag<<<cdiv(NH * NH, 256), 256, 0, stream>>>(gw3, gw3h, NH, NH);

  // --- Encoder: conv1 (s2) -> conv2 (s1) -> conv3 (s2), ego then neighbors ---
  dim3 blk(32);
  conv3x3_relu_wmma<16, 4><<<dim3(cdiv(HW1, 16), O1p / 64, 16), blk, 0, stream>>>(
      lid, wp1e, eb1, act1, (float*)nullptr, H0, W0, H1, W1, C1, 2, K1p / 32, 0);
  conv3x3_relu_wmma<16, 4><<<dim3(cdiv(HW1, 16), O1p / 64, 48), blk, 0, stream>>>(
      lid + (size_t)16 * HW0 * C0p, wp1o, ob1, act1, (float*)nullptr,
      H0, W0, H1, W1, C1, 2, K1p / 32, 16);

  conv3x3_relu_wmma<128, 4><<<dim3(cdiv(HW2, 16), O2p / 64, 16), blk, 0, stream>>>(
      act1, wp2e, eb2, act2, (float*)nullptr, H1, W1, H2, W2, C2, 1, K2 / 32, 0);
  conv3x3_relu_wmma<128, 4><<<dim3(cdiv(HW2, 16), O2p / 64, 48), blk, 0, stream>>>(
      act1 + (size_t)16 * HW1 * C1, wp2o, ob2, act2, (float*)nullptr,
      H1, W1, H2, W2, C2, 1, K2 / 32, 16);

  conv3x3_relu_wmma<64, 1><<<dim3(cdiv(HW3, 16), O3p / 16, 16), blk, 0, stream>>>(
      act2, wp3e, eb3, (_Float16*)nullptr, act3, H2, W2, H3, W3, C3, 2, K3 / 32, 0);
  conv3x3_relu_wmma<64, 1><<<dim3(cdiv(HW3, 16), O3p / 16, 48), blk, 0, stream>>>(
      act2 + (size_t)16 * HW2 * C2, wp3o, ob3, (_Float16*)nullptr, act3,
      H2, W2, H3, W3, C3, 2, K3 / 32, 16);

  // --- Node features: ego rows 0..15 (straight copy), warped neighbors 16..63 ---
  cvt_f32_f16<<<cdiv(16 * FEAT, 256), 256, 0, stream>>>(act3, xn, (long)16 * FEAT);
  warp_sample<<<cdiv(16 * 3 * HW3, 256), 256, 0, stream>>>(act3, other_tf, xn);

  // --- GCN: WMMA GEMM + (norm==0.25) max-aggregate + bias (+ReLU) ---
  gemm_wmma<4><<<dim3(4, NH / 64), blk, 0, stream>>>(xn, gw1h, xw, 64, FEAT, NH);
  gcn_combine<<<16, 256, 0, stream>>>(xw, gb1, a16, (float*)nullptr, 1);
  gemm_wmma<4><<<dim3(4, NH / 64), blk, 0, stream>>>(a16, gw2h, xw, 64, NH, NH);
  gcn_combine<<<16, 256, 0, stream>>>(xw, gb2, a16, (float*)nullptr, 1);
  gemm_wmma<4><<<dim3(4, NH / 64), blk, 0, stream>>>(a16, gw3h, xw, 64, NH, NH);
  gcn_combine<<<16, 256, 0, stream>>>(xw, gb3, (_Float16*)nullptr, pooled, 0);

  // --- Head MLP -> d_out (48 floats: pc[:,0] | pc[:,1] | pc[:,2]) ---
  head_mlp<<<1, 256, 0, stream>>>(pooled, ego_speed, mw1, mb1, mw2, mb2,
                                  fw1, fb1, fw2, fb2, fw3, fb3, (float*)d_out);
}
